// AtariGRU_17119739642082
// MI455X (gfx1250) — matrix operand order repaired and checked by reading, not compiled, not run on hardware
//
#include <hip/hip_runtime.h>
#include <stdint.h>

// Problem constants (from reference): B=256, T=128, D=512, UNITS=512
#define B_    256
#define T_    128
#define D_    512
#define NU    512
#define G3    1536      // 3*UNITS
#define MEMSZ 16
#define HSTR  516       // fp32 h-image row stride (floats)
#define BSTR  520       // bf16 LDS row stride (elems); 1040B = 260 words = 4 mod 64 banks

#if defined(__has_builtin)
#  if __has_builtin(__builtin_amdgcn_global_load_async_to_lds_b128)
#    define HAVE_ASYNC_LDS 1
#  endif
#endif
#ifndef HAVE_ASYNC_LDS
#  define HAVE_ASYNC_LDS 0
#endif

typedef __bf16 bf16_t;
typedef bf16_t   v16bf __attribute__((ext_vector_type(16)));
typedef float    v8f   __attribute__((ext_vector_type(8)));
typedef uint32_t u32x4 __attribute__((ext_vector_type(4)));

#if HAVE_ASYNC_LDS
// Builtin signature (from clang): (v4i addrspace(1)*, v4i addrspace(3)*, Ii, Ii)
typedef int v4i_t __attribute__((vector_size(16)));
typedef __attribute__((address_space(1))) v4i_t* gv4i_p;
typedef __attribute__((address_space(3))) v4i_t* lv4i_p;
#endif

union Frag { uint32_t u[8]; u32x4 q[2]; v16bf v; };

__device__ __forceinline__ uint32_t pack_bf16x2(float a, float b) {
  uint16_t lo = __builtin_bit_cast(uint16_t, (bf16_t)a);
  uint16_t hi = __builtin_bit_cast(uint16_t, (bf16_t)b);
  return (uint32_t)lo | ((uint32_t)hi << 16);
}

// A-fragment from bf16 LDS (ISA 7.12.2 16-bit A layout). Per lane the needed
// K values are contiguous: K = kb + hi*8 + [0..7] and + [16..23]  ->
// exactly two ds_load_b128, no conversion VALU.
__device__ __forceinline__ v16bf load_a_frag(const bf16_t* base) {
  Frag fa;
  fa.q[0] = *(const u32x4*)(base);
  fa.q[1] = *(const u32x4*)(base + 16);
  return fa.v;
}

// B-fragment: pre-packed lane-layout, two global_load_b128 per lane.
__device__ __forceinline__ v16bf load_b_frag(const uint32_t* __restrict__ wu,
                                             int kt, int nt, int lane) {
  const uint32_t* p = wu + (((kt * 96) + nt) << 8) + (lane << 3);
  Frag fb;
  fb.q[0] = *(const u32x4*)p;
  fb.q[1] = *(const u32x4*)(p + 4);
  return fb.v;
}

#define WMMA_BF16(A, Bv, C) \
  __builtin_amdgcn_wmma_f32_16x16x32_bf16(false, (A), false, (Bv), (short)0, (C), false, false)

// ---------------------------------------------------------------------------
// Kernel 0a: pack [W;U] (1024 x 1536 fp32) into bf16 B-fragments.
// ---------------------------------------------------------------------------
__global__ void __launch_bounds__(256) k_pack_wu(const float* __restrict__ W,
                                                 const float* __restrict__ U,
                                                 uint32_t* __restrict__ wu) {
  int o    = blockIdx.x * 256 + threadIdx.x;   // 0 .. 786431 (one u32 each)
  int f    = o >> 8;
  int slot = o & 255;
  int kt   = f / 96, nt = f % 96;
  int lane = slot >> 3, v = slot & 7;
  int kL   = ((lane >> 4) << 3) + 2 * v + ((v >> 2) << 3);  // 0..30 even
  int n    = lane & 15;
  int Kg   = kt * 32 + kL;                     // [0,512)=W rows, [512,1024)=U rows
  int Ng   = nt * 16 + n;
  const float* src = (Kg < 512) ? (W + (size_t)Kg * G3)
                                : (U + (size_t)(Kg - 512) * G3);
  wu[o] = pack_bf16x2(src[Ng], src[G3 + Ng]);
}

// ---------------------------------------------------------------------------
// Kernel 0b: one-shot fp32 -> bf16 conversion of the whole input tensor.
// ---------------------------------------------------------------------------
__global__ void __launch_bounds__(256) k_cvt_inputs(const float* __restrict__ x,
                                                    uint32_t* __restrict__ xb) {
  size_t o = (size_t)blockIdx.x * 256 + threadIdx.x;   // 8388608 u32
  float2 v = *(const float2*)(x + o * 2);
  xb[o] = pack_bf16x2(v.x, v.y);
}

// ---------------------------------------------------------------------------
// Kernel 1: segment worklist. Thread b scans row b of dones with the step
// counter, emitting reset-delimited segments (len <= 16 by MEMORY_SIZE),
// packs them densely via an LDS prefix-sum, and writes final step_f.
// entry = b<<16 | t0<<8 | len<<1 | endsWithReset
// ---------------------------------------------------------------------------
__global__ void __launch_bounds__(256) k_segments(const int* __restrict__ dones,
                                                  const int* __restrict__ step0,
                                                  uint32_t* __restrict__ entries,
                                                  uint32_t* __restrict__ total,
                                                  int* __restrict__ stepOut) {
  __shared__ int sc[256];
  const int b = threadIdx.x;
  const int* dr = dones + b * T_;

  int s = step0[b], cnt = 0, segStart = 0;
  for (int t = 0; t < T_; ++t) {
    s += 1;
    bool reset = (dr[t] == 1) || ((s % MEMSZ) == 0);
    if (reset) { cnt++; s = 0; segStart = t + 1; }
  }
  if (segStart < T_) cnt++;
  stepOut[b] = s;

  sc[b] = cnt;
  __syncthreads();
  for (int off = 1; off < 256; off <<= 1) {     // inclusive scan
    int vv = (b >= off) ? sc[b - off] : 0;
    __syncthreads();
    sc[b] += vv;
    __syncthreads();
  }
  int base = sc[b] - cnt;
  if (b == 255) *total = (uint32_t)sc[255];

  s = step0[b]; segStart = 0; int idx = base;
  for (int t = 0; t < T_; ++t) {
    s += 1;
    bool reset = (dr[t] == 1) || ((s % MEMSZ) == 0);
    if (reset) {
      int len = t - segStart + 1;
      entries[idx++] = ((uint32_t)b << 16) | ((uint32_t)segStart << 8) |
                       ((uint32_t)len << 1) | 1u;
      s = 0; segStart = t + 1;
    }
  }
  if (segStart < T_) {
    int len = T_ - segStart;
    entries[idx++] = ((uint32_t)b << 16) | ((uint32_t)segStart << 8) |
                     ((uint32_t)len << 1);
  }
}

// ---------------------------------------------------------------------------
// Kernel 2: persistent segment-tile GRU. Each WG owns 16 segments (one WMMA
// M-tile), iterates <=16 steps. Per step: fused [X|h](16x1024) @ [W;U]
// (1024x1536) via v_wmma_f32_16x16x32_bf16. A-operands come from bf16 LDS
// (2x ds_load_b128 each); h carry stays fp32 (single-owner LDS image) with a
// double-buffered bf16 mirror for the WMMA A path. X staged by async DMA.
// ---------------------------------------------------------------------------
__global__ void __launch_bounds__(256) k_gru_segments(
    const bf16_t* __restrict__ inputsBf,   // pre-converted [B,T,D] bf16
    const float* __restrict__ state0,
    const float* __restrict__ bias,        // [2][1536]
    const uint32_t* __restrict__ wu,       // packed B fragments
    const uint32_t* __restrict__ entries,
    const uint32_t* __restrict__ totalPtr,
    float* __restrict__ seqOut,
    float* __restrict__ stateOut) {
  extern __shared__ char smem[];
  float*  Hf  = (float*)smem;                         // [16][HSTR] fp32 h (single)
  bf16_t* Xb  = (bf16_t*)(smem + 33024);              // [16][BSTR] x_t bf16
  bf16_t* Hb0 = (bf16_t*)(smem + 33024 + 16640);      // [16][BSTR] h bf16 (buf A)
  bf16_t* Hb1 = (bf16_t*)(smem + 33024 + 2 * 16640);  // [16][BSTR] h bf16 (buf B)
  int* eb   = (int*)(smem + 33024 + 3 * 16640);       // [16]
  int* et0  = eb + 16;
  int* elen = et0 + 16;
  int* eend = elen + 16;
  int* mlen = eend + 16;

  const int total = (int)*totalPtr;
  const int tile  = blockIdx.x;
  if (tile * 16 >= total) return;         // padding tile (uniform exit)

  const int tid = threadIdx.x;
  if (tid < 16) {
    int g = tile * 16 + tid;
    uint32_t e = (g < total) ? entries[g] : 0u;   // pad: len = 0
    eb[tid]   = (int)(e >> 16);
    et0[tid]  = (int)((e >> 8) & 0xFF);
    elen[tid] = (int)((e >> 1) & 0x3F);
    eend[tid] = (int)(e & 1);
  }
  __syncthreads();
  if (tid == 0) {
    int m = 0;
    for (int i = 0; i < 16; ++i) m = (elen[i] > m) ? elen[i] : m;
    *mlen = m;
  }
  // init h: t0==0 segments inherit `state`, others start from reset (0)
  for (int i = tid; i < 16 * NU; i += 256) {
    int row = i >> 9, u = i & (NU - 1);
    float h0 = 0.f;
    if (elen[row] > 0 && et0[row] == 0) h0 = state0[eb[row] * NU + u];
    Hf[row * HSTR + u]  = h0;
    Hb0[row * BSTR + u] = (bf16_t)h0;
  }
  __syncthreads();

  const int maxLen = *mlen;
  const int w = tid >> 5, lane = tid & 31;
  const int laneN = lane & 15;
  const int laneHi = lane >> 4;

  bf16_t* Hbc = Hb0;
  bf16_t* Hbn = Hb1;
  const v8f vzero = {0.f, 0.f, 0.f, 0.f, 0.f, 0.f, 0.f, 0.f};

  for (int k = 0; k < maxLen; ++k) {
    // ---- stage x_t rows for active segments (bf16, pure byte copy) ----
    {
      int row = tid >> 4, seg = tid & 15;            // 16 threads x 64B per row
      if (k < elen[row]) {
        const bf16_t* src = inputsBf +
            ((size_t)eb[row] * T_ + (size_t)(et0[row] + k)) * D_ + seg * 32;
        bf16_t* dst = Xb + row * BSTR + seg * 32;
#if HAVE_ASYNC_LDS
        #pragma unroll
        for (int q = 0; q < 4; ++q)
          __builtin_amdgcn_global_load_async_to_lds_b128(
              (gv4i_p)(src + q * 8), (lv4i_p)(dst + q * 8), 0, 0);
#else
        #pragma unroll
        for (int q = 0; q < 4; ++q)
          *(u32x4*)(dst + q * 8) = *(const u32x4*)(src + q * 8);
#endif
      }
    }
#if HAVE_ASYNC_LDS
#  if __has_builtin(__builtin_amdgcn_s_wait_asynccnt)
    __builtin_amdgcn_s_wait_asynccnt(0);
#  else
    asm volatile("s_wait_asynccnt 0x0" ::: "memory");
#  endif
#endif
    __syncthreads();

    // ---- GEMM + epilogue, 2 unit-tiles per group, 2 groups per wave ----
    #pragma unroll 1
    for (int g = 0; g < 2; ++g) {
      const int ugBase = (w << 2) + (g << 1);   // unit-tile pair [ugBase, ugBase+1]
      v8f zA0 = vzero, zA1 = vzero;
      v8f rA0 = vzero, rA1 = vzero;
      v8f xA0 = vzero, xA1 = vzero;
      v8f hA0 = vzero, hA1 = vzero;

      const bf16_t* aRowX = Xb + laneN * BSTR + (laneHi << 3);
      const bf16_t* aRowH = Hbc + laneN * BSTR + (laneHi << 3);

      // pass 1: K = 0..511 (A = x_t), accumulate z, r, xh
      #pragma unroll 2
      for (int kt = 0; kt < 16; ++kt) {
        v16bf a = load_a_frag(aRowX + (kt << 5));
        v16bf bz0 = load_b_frag(wu, kt, ugBase,      lane);
        v16bf br0 = load_b_frag(wu, kt, 32 + ugBase, lane);
        v16bf bx0 = load_b_frag(wu, kt, 64 + ugBase, lane);
        v16bf bz1 = load_b_frag(wu, kt, ugBase + 1,  lane);
        v16bf br1 = load_b_frag(wu, kt, 33 + ugBase, lane);
        v16bf bx1 = load_b_frag(wu, kt, 65 + ugBase, lane);
        zA0 = WMMA_BF16(a, bz0, zA0);
        rA0 = WMMA_BF16(a, br0, rA0);
        xA0 = WMMA_BF16(a, bx0, xA0);
        zA1 = WMMA_BF16(a, bz1, zA1);
        rA1 = WMMA_BF16(a, br1, rA1);
        xA1 = WMMA_BF16(a, bx1, xA1);
      }
      // pass 2: K = 512..1023 (A = h), accumulate z, r, hh
      #pragma unroll 2
      for (int kt = 16; kt < 32; ++kt) {
        v16bf a = load_a_frag(aRowH + ((kt - 16) << 5));
        v16bf bz0 = load_b_frag(wu, kt, ugBase,      lane);
        v16bf br0 = load_b_frag(wu, kt, 32 + ugBase, lane);
        v16bf bh0 = load_b_frag(wu, kt, 64 + ugBase, lane);
        v16bf bz1 = load_b_frag(wu, kt, ugBase + 1,  lane);
        v16bf br1 = load_b_frag(wu, kt, 33 + ugBase, lane);
        v16bf bh1 = load_b_frag(wu, kt, 65 + ugBase, lane);
        zA0 = WMMA_BF16(a, bz0, zA0);
        rA0 = WMMA_BF16(a, br0, rA0);
        hA0 = WMMA_BF16(a, bh0, hA0);
        zA1 = WMMA_BF16(a, bz1, zA1);
        rA1 = WMMA_BF16(a, br1, rA1);
        hA1 = WMMA_BF16(a, bh1, hA1);
      }

      // ---- fp32 GRU epilogue on C layout (M = i + 8*laneHi, N = laneN) ----
      #pragma unroll
      for (int j = 0; j < 2; ++j) {
        const v8f& zz = j ? zA1 : zA0;
        const v8f& rr = j ? rA1 : rA0;
        const v8f& xx = j ? xA1 : xA0;
        const v8f& hh = j ? hA1 : hA0;
        int u  = ((ugBase + j) << 4) + laneN;
        float bz  = bias[u]        + bias[G3 + u];
        float br  = bias[512 + u]  + bias[G3 + 512 + u];
        float bxh = bias[1024 + u];
        float bhh = bias[G3 + 1024 + u];
        #pragma unroll
        for (int i = 0; i < 8; ++i) {
          int M = i + (laneHi << 3);
          float z  = 1.f / (1.f + __expf(-(zz[i] + bz)));
          float r  = 1.f / (1.f + __expf(-(rr[i] + br)));
          float hc = tanhf(xx[i] + bxh + r * (hh[i] + bhh));
          float hp = Hf[M * HSTR + u];          // single-owner read
          float hn = z * hp + (1.f - z) * hc;
          Hf[M * HSTR + u]  = hn;               // single-owner write
          Hbn[M * BSTR + u] = (bf16_t)hn;       // bf16 mirror for next GEMM
          int len = elen[M];
          if (k < len) {
            int bb = eb[M];
            int t  = et0[M] + k;
            seqOut[((size_t)bb * T_ + t) * NU + u] = hn;   // pre-reset output
            if (t == T_ - 1 && k == len - 1)
              stateOut[bb * NU + u] = eend[M] ? 0.f : hn;  // post-reset state
          }
        }
      }
    }
    __syncthreads();
    bf16_t* tmp = Hbc; Hbc = Hbn; Hbn = tmp;
  }
}

// ---------------------------------------------------------------------------
extern "C" void kernel_launch(void* const* d_in, const int* in_sizes, int n_in,
                              void* d_out, int out_size, void* d_ws, size_t ws_size,
                              hipStream_t stream) {
  const float* inputs = (const float*)d_in[0];
  const int*   dones  = (const int*)d_in[1];
  const float* state0 = (const float*)d_in[2];
  const int*   step0  = (const int*)d_in[3];
  const float* W      = (const float*)d_in[4];
  const float* U      = (const float*)d_in[5];
  const float* bias   = (const float*)d_in[6];

  // workspace: 3 MB packed weights + 128 KB worklist + 32 MB bf16 inputs
  uint32_t* wu       = (uint32_t*)d_ws;               // 786432 u32
  uint32_t* entries  = wu + 786432;                   // 32768 u32
  uint32_t* total    = entries + 32768;               // 1 u32
  bf16_t*   inputsBf = (bf16_t*)((char*)d_ws + (4u << 20));  // 16777216 bf16

  float* seqOut   = (float*)d_out;                          // [B,T,U]
  float* stateOut = seqOut + (size_t)B_ * T_ * NU;          // [B,U]
  int*   stepOut  = (int*)(stateOut + (size_t)B_ * NU);     // [B,1] int32 bits

  k_pack_wu<<<3072, 256, 0, stream>>>(W, U, wu);
  k_cvt_inputs<<<32768, 256, 0, stream>>>(inputs, (uint32_t*)inputsBf);
  k_segments<<<1, 256, 0, stream>>>(dones, step0, entries, total, stepOut);

  size_t shmem = 33024 + 3 * 16640 + 68 * sizeof(int);      // ~83 KB
  k_gru_segments<<<2048, 256, shmem, stream>>>(inputsBf, state0, bias, wu,
                                               entries, total, seqOut, stateOut);
}
